// AttentionLayer_46789373722946
// MI455X (gfx1250) — compile-verified, matrix operand order checked by reading, not run on hardware
//
#include <hip/hip_runtime.h>

// Problem constants (from reference): B=32, T=4096, H=256
#define BSZ   32
#define TSZ   4096
#define HSZ   256
#define LPAD  264          // padded LDS row (halfs): 264*2=528B = 16B-aligned rows
#define TCH   512          // T-chunk for weighted sum
#define NCH   (TSZ / TCH)  // 8 chunks

typedef __attribute__((ext_vector_type(16))) _Float16     v16h;
typedef __attribute__((ext_vector_type(8)))  _Float16     v8h;
typedef __attribute__((ext_vector_type(8)))  float        v8f;
typedef __attribute__((ext_vector_type(4)))  unsigned int v4u;
typedef __attribute__((ext_vector_type(8)))  int          v8i;
typedef __attribute__((ext_vector_type(4)))  int          v4i;

// Load 16 halfs from a 16B-aligned LDS/global address as two aligned v8h.
static __device__ inline v16h load16h(const _Float16* p) {
    v8h lo = *(const v8h*)(p);
    v8h hi = *(const v8h*)(p + 8);
    return __builtin_shufflevector(lo, hi, 0,1,2,3,4,5,6,7,8,9,10,11,12,13,14,15);
}

// Branch-free tanh: 1 - 2/(exp(2x)+1).  Native v_exp_f32 + v_rcp_f32,
// no EXEC divergence, graceful at +/-overflow (-> +/-1), no NaN.
static __device__ inline float fast_tanh(float x) {
    const float e = __expf(x + x);
    return 1.0f - 2.0f * __builtin_amdgcn_rcpf(e + 1.0f);
}

// ---------------------------------------------------------------------------
// Kernel 0 (one-shot): weight f32 [h][n]  ->  f16 transposed wt16[n][h].
// 64K elements; runs once, result stays L2-resident for all GEMM blocks.
// ---------------------------------------------------------------------------
__global__ __launch_bounds__(256) void weight_to_f16T_kernel(
    const float* __restrict__ weight, _Float16* __restrict__ wt16)
{
    const int n = blockIdx.x;     // weight column = wt16 row
    const int h = threadIdx.x;    // weight row (K dim)
    wt16[n * HSZ + h] = (_Float16)weight[h * HSZ + n];
}

// ---------------------------------------------------------------------------
// Kernel 1: fused  logits[b,t] = (sum_n tanh((x W)[n] + bias[n]) * proj[n])/16
// masked to -1e6.  GEMM via v_wmma_f32_16x16x32_f16; weight^T(f16) staged
// into LDS by the Tensor Data Mover with hardware row padding.
// ---------------------------------------------------------------------------
__global__ __launch_bounds__(256) void attn_logits_kernel(
    const float* __restrict__ xes, const unsigned char* __restrict__ mask,
    const _Float16* __restrict__ wt16, const float* __restrict__ bias,
    const float* __restrict__ proj, float* __restrict__ logits)
{
    extern __shared__ _Float16 lds[];
    _Float16* WtT   = lds;                 // [HSZ][LPAD]  WtT[n][h] = weight[h][n]
    _Float16* xsAll = lds + HSZ * LPAD;    // [8 waves][16 rows][LPAD]

    const int tid  = threadIdx.x;
    const int lane = tid & 31;
    const int wave = tid >> 5;
    const int lr   = lane & 15;            // A: row-in-tile  / B,C: n-in-tile
    const int kh   = (lane >> 4) << 4;     // K half-offset: 0 or 16

    // --- TDM: DMA the whole 256x256 f16 weight^T tile into LDS. ------------
    // D# (CDNA5 ISA): 2-D tile, data_size=2B, tensor/tile dims
    // 256x256, dim0 stride 256.  pad_enable with pad_interval=6 (128 DWORDs
    // = one 512B row) and pad_amount=3 (4 DWORDs = 16B) gives LDS row pitch
    // 528B = LPAD halfs -- the hardware does the padded layout for us.
    if (wave == 0) {
        const unsigned long long ga = (unsigned long long)(uintptr_t)wt16;
        v4u g0;
        g0.x = 1u;                                    // count=1, user descriptor
        g0.y = 0u;                                    // lds_addr: dyn-LDS base (WtT)
        g0.z = (unsigned)ga;                          // global_addr[31:0]
        g0.w = (unsigned)((ga >> 32) & 0x01FFFFFFu)   // global_addr[56:32]
             | (2u << 30);                            // type = 2 ("image")
        v8i g1;
        g1[0] = (1 << 16)                             // data_size = 2 bytes
              | (1 << 20)                             // pad_enable
              | (6 << 22)                             // pad_interval: 2^7=128 DW
              | (3 << 25);                            // pad_amount: 4 DW
        g1[1] = (int)((HSZ & 0xffff) << 16);          // tensor_dim0[15:0] @ [63:48]
        g1[2] = (int)(((HSZ >> 16) & 0xffff)          // tensor_dim0[31:16]
              | ((HSZ & 0xffff) << 16));              // tensor_dim1[15:0]
        g1[3] = (int)(((HSZ >> 16) & 0xffff)          // tensor_dim1[31:16]
              | ((HSZ & 0xffff) << 16));              // tile_dim0 = 256
        g1[4] = HSZ;                                  // tile_dim1=256 | tile_dim2=0
        g1[5] = HSZ;                                  // tensor_dim0_stride[31:0]
        g1[6] = 0;                                    // stride0 hi | dim1_stride lo
        g1[7] = 0;                                    // dim1_stride hi
        const v4i gz4 = {0, 0, 0, 0};                 // groups 2/3 unused (2-D)
        const v8i gz8 = {0, 0, 0, 0, 0, 0, 0, 0};
        __builtin_amdgcn_tensor_load_to_lds(g0, g1, gz4, gz4, gz8, 0);
    }

    // Stage this wave's 16 time-rows of x, f32 -> f16 (overlaps the TDM).
    const int rowBase = (blockIdx.x * 8 + wave) * 16;   // flattened b*T + t
    _Float16* xs = xsAll + wave * 16 * LPAD;
    for (int i = lane; i < 16 * HSZ; i += 32) {
        const int r = i >> 8, k = i & 255;
        xs[r * LPAD + k] = (_Float16)xes[(rowBase + r) * HSZ + k];
    }

    if (wave == 0) __builtin_amdgcn_s_wait_tensorcnt(0);  // TDM done -> LDS valid
    __syncthreads();                                      // publish to all waves

    // A fragments for the full K=256 strip (8 x K32 tiles) kept in VGPRs.
    v16h afrag[8];
#pragma unroll
    for (int kt = 0; kt < 8; ++kt)
        afrag[kt] = load16h(xs + lr * LPAD + kt * 32 + kh);

    float rowacc[8] = {0.f,0.f,0.f,0.f,0.f,0.f,0.f,0.f};

#pragma unroll 4
    for (int nt = 0; nt < 16; ++nt) {
        const _Float16* wb = WtT + (nt * 16 + lr) * LPAD + kh;
        v8f acc = {};
        // One-deep software pipeline: fetch b[kt+1] while wmma consumes b[kt].
        v16h bcur = load16h(wb);
#pragma unroll
        for (int kt = 0; kt < 8; ++kt) {
            v16h bnext = bcur;
            if (kt < 7) bnext = load16h(wb + (kt + 1) * 32);
            acc = __builtin_amdgcn_wmma_f32_16x16x32_f16(
                      false, afrag[kt], false, bcur,
                      (short)0, acc, false, false);
            bcur = bnext;
        }
        // Epilogue: tanh + bias, dot with proj over this N-tile (branch-free).
        const int   n  = nt * 16 + lr;
        const float bn = bias[n];
        const float pn = proj[n];
#pragma unroll
        for (int i = 0; i < 8; ++i)
            rowacc[i] += fast_tanh(acc[i] + bn) * pn;
    }

    // Reduce over N (distinct n live in lanes of the same 16-lane half).
#pragma unroll
    for (int i = 0; i < 8; ++i) {
        float v = rowacc[i];
        v += __shfl_xor(v, 8, 32);
        v += __shfl_xor(v, 4, 32);
        v += __shfl_xor(v, 2, 32);
        v += __shfl_xor(v, 1, 32);
        rowacc[i] = v;
    }
    if (lr == 0) {
        const int rOff = (lane >> 4) * 8;  // lane 0 -> rows 0..7, lane 16 -> 8..15
#pragma unroll
        for (int i = 0; i < 8; ++i) {
            const int r  = rowBase + rOff + i;
            float     lg = rowacc[i] * (1.0f / 16.0f);
            if (!mask[r]) lg = -1000000.0f;
            logits[r] = lg;
        }
    }
}

// ---------------------------------------------------------------------------
// Kernel 2: per-batch softmax stats (max, sum of exp) over T=4096.
// ---------------------------------------------------------------------------
__global__ __launch_bounds__(256) void softmax_stats_kernel(
    const float* __restrict__ logits, float* __restrict__ stats)
{
    __shared__ float red[256];
    const int b = blockIdx.x, tid = threadIdx.x;
    const float* lg = logits + b * TSZ;

    float m = -3.4e38f;
    for (int t = tid; t < TSZ; t += 256) m = fmaxf(m, lg[t]);
    red[tid] = m; __syncthreads();
    for (int s = 128; s > 0; s >>= 1) {
        if (tid < s) red[tid] = fmaxf(red[tid], red[tid + s]);
        __syncthreads();
    }
    m = red[0]; __syncthreads();

    float sum = 0.f;
    for (int t = tid; t < TSZ; t += 256) sum += __expf(lg[t] - m);
    red[tid] = sum; __syncthreads();
    for (int s = 128; s > 0; s >>= 1) {
        if (tid < s) red[tid] += red[tid + s];
        __syncthreads();
    }
    if (tid == 0) { stats[b * 2] = m; stats[b * 2 + 1] = red[0]; }
}

// ---------------------------------------------------------------------------
// Kernel 3: partial weighted sums  partials[b,c,h] = sum_{t in chunk} w[t]*x[b,t,h]
// ---------------------------------------------------------------------------
__global__ __launch_bounds__(256) void weighted_partial_kernel(
    const float* __restrict__ xes, const float* __restrict__ logits,
    const float* __restrict__ stats, float* __restrict__ partials)
{
    __shared__ float w[TCH];
    const int b = blockIdx.x, c = blockIdx.y, h = threadIdx.x;
    const float m   = stats[b * 2];
    const float inv = 1.0f / stats[b * 2 + 1];
    const int   t0  = c * TCH;

    for (int i = h; i < TCH; i += 256)
        w[i] = __expf(logits[b * TSZ + t0 + i] - m) * inv;
    __syncthreads();

    float acc = 0.f;
    const float* xp = xes + ((size_t)b * TSZ + t0) * HSZ + h;
    for (int i = 0; i < TCH; ++i)
        acc += w[i] * xp[(size_t)i * HSZ];

    partials[((b * NCH) + c) * HSZ + h] = acc;
}

// ---------------------------------------------------------------------------
// Kernel 4: reduce chunk partials -> out[b,h]
// ---------------------------------------------------------------------------
__global__ __launch_bounds__(256) void reduce_partials_kernel(
    const float* __restrict__ partials, float* __restrict__ out)
{
    const int b = blockIdx.x, h = threadIdx.x;
    float acc = 0.f;
#pragma unroll
    for (int c = 0; c < NCH; ++c)
        acc += partials[((b * NCH) + c) * HSZ + h];
    out[b * HSZ + h] = acc;
}

// ---------------------------------------------------------------------------
extern "C" void kernel_launch(void* const* d_in, const int* in_sizes, int n_in,
                              void* d_out, int out_size, void* d_ws, size_t ws_size,
                              hipStream_t stream) {
    (void)in_sizes; (void)n_in; (void)out_size; (void)ws_size;
    const float*         xes    = (const float*)d_in[0];
    const unsigned char* mask   = (const unsigned char*)d_in[1];  // jnp bool: 1 byte
    const float*         weight = (const float*)d_in[2];
    const float*         bias   = (const float*)d_in[3];
    const float*         proj   = (const float*)d_in[4];
    float*               out    = (float*)d_out;

    // Workspace layout: logits[B*T] f32 | stats[64] f32 | partials[B*NCH*H] f32
    //                   | wt16[H*H] f16
    float* logits   = (float*)d_ws;
    float* stats    = logits + (size_t)BSZ * TSZ;
    float* partials = stats + 64;
    _Float16* wt16  = (_Float16*)(partials + (size_t)BSZ * NCH * HSZ);

    const size_t shmem = (size_t)(HSZ * LPAD + 8 * 16 * LPAD) * sizeof(_Float16);

    weight_to_f16T_kernel<<<dim3(HSZ), 256, 0, stream>>>(weight, wt16);
    attn_logits_kernel<<<dim3((BSZ * TSZ) / 128), 256, shmem, stream>>>(
        xes, mask, wt16, bias, proj, logits);
    softmax_stats_kernel<<<dim3(BSZ), 256, 0, stream>>>(logits, stats);
    weighted_partial_kernel<<<dim3(BSZ, NCH), 256, 0, stream>>>(
        xes, logits, stats, partials);
    reduce_partials_kernel<<<dim3(BSZ), 256, 0, stream>>>(partials, out);
}
